// BiF_AGRU_LSTM_18073222381853
// MI455X (gfx1250) — matrix operand order repaired and checked by reading, not compiled, not run on hardware
//
#include <hip/hip_runtime.h>
#include <hip/hip_bf16.h>
#include <math.h>

// ---------------------------------------------------------------------------
// Problem constants (from the reference)
// ---------------------------------------------------------------------------
#define BB   512     // batch (utterances)
#define SS   100     // sequence length
#define DD   300     // embedding dim
#define HH   300     // LSTM hidden
#define DP   320     // D/H padded to multiple of 32 (WMMA K granularity)
#define G4   1200    // 4*H LSTM gate width
#define KC2  640     // concatenated LSTM K: [Wih | Whh]
#define CAT  608     // 2*H=600 padded to multiple of 32
#define DA   100
#define DAP  128     // DA padded to multiple of 32
#define G3   300     // 3*DA GRU gate width
#define G3P  304     // padded to multiple of 16 (N tiles)
#define W1   40      // context window
#define QQ   511     // B-1 queries
#define NHOPS 3
#define NC   7
#define LIN_NP 112   // lin1 N=100 padded to 16
#define ATT_NP 112

typedef __bf16 bf16_t;
typedef __attribute__((ext_vector_type(16))) __bf16 v16bf;
typedef __attribute__((ext_vector_type(8)))  __bf16 v8bf;
typedef __attribute__((ext_vector_type(8)))  float  v8f;

__device__ __forceinline__ float sigf(float x) { return 1.0f / (1.0f + __expf(-x)); }

// Async global->LDS path (ASYNCcnt, CDNA5 ISA), guarded.
#if defined(__has_builtin)
#  if __has_builtin(__builtin_amdgcn_global_load_async_to_lds_b128)
#    define ASYNC_LDS 1
#  endif
#endif
#ifndef ASYNC_LDS
#  define ASYNC_LDS 0
#endif

#if defined(__has_builtin)
#  if __has_builtin(__builtin_amdgcn_s_wait_asynccnt)
#    define WAIT_ASYNC0() __builtin_amdgcn_s_wait_asynccnt(0)
#  endif
#endif
#ifndef WAIT_ASYNC0
#  define WAIT_ASYNC0() asm volatile("s_wait_asynccnt 0x0" ::: "memory")
#endif

#if ASYNC_LDS
// Builtin signature (from compiler diagnostic): (v4i AS1*, v4i AS3*, Imm, Imm)
typedef int v4i __attribute__((vector_size(16)));
typedef __attribute__((address_space(1))) v4i as1_v4i;
typedef __attribute__((address_space(3))) v4i as3_v4i;
#define ASYNC_CP_B128(gsrc, ldst)                                         \
    __builtin_amdgcn_global_load_async_to_lds_b128(                       \
        (as1_v4i*)(uintptr_t)(gsrc), (as3_v4i*)(ldst), 0, 0)
#endif

// ---------------------------------------------------------------------------
// bf16 WMMA GEMM, dual-A-source (K-concatenated), 16x64 per wave, LDS-shared B.
//   C[M,N] = concat_K(A1,A2) * Bw^T (+bias) (tanh?)
// A1/A2: row-major bf16, K1/K2 multiples of 32 (zero-padded cols).
// Bw: row-major [nRowsPad, ldb] bf16 (N rows padded, K = K1+K2 cols padded).
// Block = 128 threads = 4 waves covering 64 rows x 64 cols; the 64-col B strip
// is staged in LDS (double-buffered async fill when available: one barrier per
// K chunk, fill of chunk i+1 overlaps WMMAs of chunk i).
// Fragments (bf16 A 16x32 / B 32x16) use contiguous k = 16*(lane/16) + s in both A and B.
// ---------------------------------------------------------------------------
__global__ __launch_bounds__(128)
void k_gemm_bf16(const bf16_t* __restrict__ A1, int lda1, int K1,
                 const bf16_t* __restrict__ A2, int lda2, int K2,
                 const bf16_t* __restrict__ Bw, int ldb, int nRowsPad,
                 float* __restrict__ C, int ldc,
                 const float* __restrict__ bias,
                 int M, int N, int Mt, int Nt4, int actFlag)
{
#if ASYNC_LDS
    __shared__ __align__(32) bf16_t Bs[2][64 * 32];   // double-buffered 4 KB strips
#else
    __shared__ __align__(32) bf16_t Bs[1][64 * 32];
#endif

    const int lane = threadIdx.x;                  // 0..31
    const int wave = threadIdx.y;                  // 0..3
    const int tid  = wave * 32 + lane;
    const int g    = lane >> 4;                    // half-wave select
    const int lr   = lane & 15;

    int mtile = (blockIdx.x / Nt4) * 4 + wave;     // clamp (no early-return:
    if (mtile >= Mt) mtile = Mt - 1;               //  all threads must barrier)
    const int nBase = (blockIdx.x % Nt4) * 64;
    const int m0 = mtile * 16;

    int arow = m0 + lr; if (arow >= M) arow = M - 1;     // stay in-bounds

    // B-strip fill: 128 threads x 16 halves (32 B) = full 64x32 tile
    const int fc  = tid >> 1;                      // column within strip 0..63
    const int fkh = (tid & 1) * 16;                // K-half 0 / 16
    int bcol = nBase + fc; if (bcol >= nRowsPad) bcol = nRowsPad - 1;
    const bf16_t* Bp = Bw + (size_t)bcol * ldb;

    v8f acc0 = {}, acc1 = {}, acc2 = {}, acc3 = {};
    const int Ktot = K1 + K2;

#if ASYNC_LDS
    {   // prologue: async fill of chunk 0 into buffer 0
        const bf16_t* src = Bp + fkh;
        ASYNC_CP_B128(src,     &Bs[0][fc * 32 + fkh]);
        ASYNC_CP_B128(src + 8, &Bs[0][fc * 32 + fkh + 8]);
    }
#endif

    for (int k0 = 0; k0 < Ktot; k0 += 32) {
        // per-wave A fragment from the active source
        const bf16_t* Ap; int kl;
        if (k0 < K1) { Ap = A1 + (size_t)arow * lda1; kl = k0; }
        else         { Ap = A2 + (size_t)arow * lda2; kl = k0 - K1; }
        v8bf alo = *(const v8bf*)(Ap + kl + g * 16);
        v8bf ahi = *(const v8bf*)(Ap + kl + g * 16 + 8);
        __builtin_prefetch(Ap + kl + 32, 0, 1);    // next K chunk (global_prefetch_b8)

#if ASYNC_LDS
        const int p = (k0 >> 5) & 1;
        WAIT_ASYNC0();                 // own chunk-i fill retired (in-order)
        __syncthreads();               // all waves: fill done AND done reading Bs[p^1]
        if (k0 + 32 < Ktot) {          // overlap: fill chunk i+1 into other buffer
            const bf16_t* src = Bp + (k0 + 32) + fkh;
            ASYNC_CP_B128(src,     &Bs[p ^ 1][fc * 32 + fkh]);
            ASYNC_CP_B128(src + 8, &Bs[p ^ 1][fc * 32 + fkh + 8]);
        }
        const bf16_t* Bt = Bs[p];
#else
        *(v16bf*)&Bs[0][fc * 32 + fkh] = *(const v16bf*)(Bp + k0 + fkh);
        __syncthreads();
        const bf16_t* Bt = Bs[0];
#endif

        v16bf av = __builtin_shufflevector(alo, ahi,
                     0,1,2,3,4,5,6,7,8,9,10,11,12,13,14,15);
        v16bf b0 = *(const v16bf*)&Bt[( 0 + lr) * 32 + g * 16];
        v16bf b1 = *(const v16bf*)&Bt[(16 + lr) * 32 + g * 16];
        v16bf b2 = *(const v16bf*)&Bt[(32 + lr) * 32 + g * 16];
        v16bf b3 = *(const v16bf*)&Bt[(48 + lr) * 32 + g * 16];
        acc0 = __builtin_amdgcn_wmma_f32_16x16x32_bf16(false, av, false, b0, (short)0, acc0, false, false);
        acc1 = __builtin_amdgcn_wmma_f32_16x16x32_bf16(false, av, false, b1, (short)0, acc1, false, false);
        acc2 = __builtin_amdgcn_wmma_f32_16x16x32_bf16(false, av, false, b2, (short)0, acc2, false, false);
        acc3 = __builtin_amdgcn_wmma_f32_16x16x32_bf16(false, av, false, b3, (short)0, acc3, false, false);

#if !ASYNC_LDS
        __syncthreads();
#endif
    }

    // epilogue: 4 sub-tiles; C layout: VGPR r -> M = m0 + (lane>>4)*8 + r
    v8f accs[4] = {acc0, acc1, acc2, acc3};
#pragma unroll
    for (int j = 0; j < 4; ++j) {
        const int col = nBase + j * 16 + lr;
#pragma unroll
        for (int r = 0; r < 8; ++r) {
            const int row = m0 + g * 8 + r;
            if (row < M && col < N) {
                float v = accs[j][r];
                if (bias)    v += bias[col];
                if (actFlag) v = tanhf(v);
                C[(size_t)row * ldc + col] = v;
            }
        }
    }
}

// ---------------------------------------------------------------------------
// Elementwise / glue kernels
// ---------------------------------------------------------------------------
__global__ void k_cast_pad(const float* __restrict__ src, bf16_t* __restrict__ dst,
                           int rows, int K, int rowsPad, int Kpad,
                           int dstLd, int colOfs)
{
    int i = blockIdx.x * blockDim.x + threadIdx.x;
    if (i >= rowsPad * Kpad) return;
    int r = i / Kpad, k = i % Kpad;
    float v = (r < rows && k < K) ? src[(size_t)r * K + k] : 0.0f;
    dst[(size_t)r * dstLd + colOfs + k] = (bf16_t)v;
}

__global__ void k_vec_add(const float* a, const float* b, float* o, int n)
{
    int i = blockIdx.x * blockDim.x + threadIdx.x;
    if (i < n) o[i] = a[i] + b[i];
}

__global__ void k_gather_embed(const int* __restrict__ sents,
                               const float* __restrict__ emb,
                               bf16_t* __restrict__ X)
{
    int i = blockIdx.x * blockDim.x + threadIdx.x;   // over SS*BB*DP
    if (i >= SS * BB * DP) return;
    int d = i % DP; int sb = i / DP; int b = sb % BB; int s = sb / BB;
    float v = 0.0f;
    if (d < DD) { int tok = sents[b * SS + s]; v = emb[(size_t)tok * DD + d]; }
    X[i] = (bf16_t)v;
}

__global__ void k_lstm_point(const float* __restrict__ Gm,
                             float* __restrict__ h, float* __restrict__ c,
                             bf16_t* __restrict__ hbf, bf16_t* __restrict__ OUT,
                             const int* __restrict__ lengths, int s, int outOfs)
{
    int i = blockIdx.x * blockDim.x + threadIdx.x;   // over BB*DP
    if (i >= BB * DP) return;
    int col = i % DP, b = i / DP;
    if (col >= HH) { hbf[b * DP + col] = (bf16_t)0.0f; return; }
    const float* g = Gm + (size_t)b * G4;
    float ig = g[col], fg = g[HH + col], gg = g[2 * HH + col], og = g[3 * HH + col];
    float cprev = c[b * HH + col], hprev = h[b * HH + col];
    float cn = sigf(fg) * cprev + sigf(ig) * tanhf(gg);
    float hn = sigf(og) * tanhf(cn);
    bool  m  = (s < lengths[b]);
    float hv = m ? hn : hprev, cv = m ? cn : cprev;
    h[b * HH + col] = hv; c[b * HH + col] = cv;
    hbf[b * DP + col] = (bf16_t)hv;
    OUT[((size_t)s * BB + b) * CAT + outOfs + col] = (bf16_t)(m ? hn : 0.0f);
}

__global__ void k_maxpool(const bf16_t* __restrict__ OUT, bf16_t* __restrict__ mp)
{
    int i = blockIdx.x * blockDim.x + threadIdx.x;   // over BB*CAT
    if (i >= BB * CAT) return;
    int j = i % CAT, b = i / CAT;
    float v = 0.0f;
    if (j < 600) {
        float mx = -1e30f;
        for (int s = 0; s < SS; ++s)
            mx = fmaxf(mx, (float)OUT[((size_t)s * BB + b) * CAT + j]);
        v = mx;                                       // pads contributed 0 already
    }
    mp[i] = (bf16_t)v;
}

__global__ void k_build_bt(const float* __restrict__ s_utt, bf16_t* __restrict__ bt)
{
    int i = blockIdx.x * blockDim.x + threadIdx.x;   // over W1*QQ*DAP
    if (i >= W1 * QQ * DAP) return;
    int d = i % DAP; int r = i / DAP; int q = r % QQ; int t = r / QQ;
    int idx = (q + 1) - W1 + t;
    float v = (d < DA && idx >= 0) ? s_utt[idx * DA + d] : 0.0f;
    bt[i] = (bf16_t)v;
}

__global__ void k_gru_point(const float* __restrict__ xg, const float* __restrict__ hg,
                            float* __restrict__ h, bf16_t* __restrict__ hbf,
                            float* __restrict__ memOut)
{
    int i = blockIdx.x * blockDim.x + threadIdx.x;   // over QQ*DAP
    if (i >= QQ * DAP) return;
    int d = i % DAP, q = i / DAP;
    if (d >= DA) { hbf[q * DAP + d] = (bf16_t)0.0f; return; }
    float xr = xg[q * G3 + d], xz = xg[q * G3 + DA + d], xn = xg[q * G3 + 2 * DA + d];
    float hr = hg[q * G3 + d], hz = hg[q * G3 + DA + d], hn = hg[q * G3 + 2 * DA + d];
    float r = sigf(xr + hr), z = sigf(xz + hz);
    float n = tanhf(xn + r * hn);
    float h2 = (1.0f - z) * n + z * h[q * DA + d];
    h[q * DA + d] = h2; hbf[q * DAP + d] = (bf16_t)h2;
    memOut[q * DA + d] = h2;
}

__global__ void k_mem_combine(const float* __restrict__ s_utt,
                              const float* __restrict__ memF, const float* __restrict__ memB,
                              float* __restrict__ memTm, bf16_t* __restrict__ membf)
{
    int i = blockIdx.x * blockDim.x + threadIdx.x;   // over W1*QQ*DAP
    if (i >= W1 * QQ * DAP) return;
    int d = i % DAP; int r = i / DAP; int q = r % QQ; int t = r / QQ;
    if (d >= DA) { membf[r * DAP + d] = (bf16_t)0.0f; return; }
    int idx = (q + 1) - W1 + t;
    float btv = (idx >= 0) ? s_utt[idx * DA + d] : 0.0f;
    float v = btv + memF[r * DA + d] + memB[r * DA + d];
    memTm[r * DA + d] = v;
    membf[r * DAP + d] = (bf16_t)v;
}

__global__ void k_copy(const float* a, float* o, int n)
{
    int i = blockIdx.x * blockDim.x + threadIdx.x;
    if (i < n) o[i] = a[i];
}

__global__ __launch_bounds__(64)
void k_attn_softmax(const float* __restrict__ eps, const float* __restrict__ memTm,
                    float* __restrict__ scores)
{
    __shared__ float sc[W1];
    __shared__ float mx, sm;
    int q = blockIdx.x, k = threadIdx.x;
    if (k < W1) {
        float dot = 0.0f;
        const float* e = eps + q * DA;
        const float* m = memTm + ((size_t)k * QQ + q) * DA;
        for (int d = 0; d < DA; ++d) dot += e[d] * m[d];
        int idx = (q + 1) - W1 + k;
        sc[k] = (idx >= 0) ? dot : -1e10f;
    }
    __syncthreads();
    if (k == 0) {
        float m0 = -1e30f;
        for (int j = 0; j < W1; ++j) m0 = fmaxf(m0, sc[j]);
        float s0 = 0.0f;
        for (int j = 0; j < W1; ++j) s0 += __expf(sc[j] - m0);
        mx = m0; sm = s0;
    }
    __syncthreads();
    if (k < W1) scores[q * W1 + k] = __expf(sc[k] - mx) / sm;
}

__global__ void k_attgru_point(const float* __restrict__ cr, const float* __restrict__ cw,
                               const float* __restrict__ hr, const float* __restrict__ hu,
                               const float* __restrict__ scores,
                               float* __restrict__ h, bf16_t* __restrict__ hbf, int t)
{
    int i = blockIdx.x * blockDim.x + threadIdx.x;   // over QQ*DAP
    if (i >= QQ * DAP) return;
    int d = i % DAP, q = i / DAP;
    if (d >= DA) { hbf[q * DAP + d] = (bf16_t)0.0f; return; }
    float g  = scores[q * W1 + t];
    size_t rw = ((size_t)t * QQ + q) * DA + d;
    float r  = sigf(cr[rw] + hr[q * DA + d]);
    float ht = tanhf(cw[rw] + r * hu[q * DA + d]);
    float h2 = g * ht + (1.0f - g) * h[q * DA + d];
    h[q * DA + d] = h2; hbf[q * DAP + d] = (bf16_t)h2;
}

__global__ void k_eps_add(float* eps, const float* h, int n)
{
    int i = blockIdx.x * blockDim.x + threadIdx.x;
    if (i < n) eps[i] += h[i];
}

__global__ __launch_bounds__(32)
void k_classify(const float* __restrict__ s_utt, const float* __restrict__ eps,
                const float* __restrict__ clsW, const float* __restrict__ clsb,
                float* __restrict__ out)
{
    __shared__ float lg[NC];
    __shared__ float mx, sm;
    int b = blockIdx.x, j = threadIdx.x;
    const float* x = (b == 0) ? s_utt : (eps + (b - 1) * DA);
    if (j < NC) {
        float d = clsb[j];
        for (int t = 0; t < DA; ++t) d += x[t] * clsW[j * DA + t];
        lg[j] = d;
    }
    __syncthreads();
    if (j == 0) {
        float m0 = -1e30f;
        for (int k = 0; k < NC; ++k) m0 = fmaxf(m0, lg[k]);
        float s0 = 0.0f;
        for (int k = 0; k < NC; ++k) s0 += __expf(lg[k] - m0);
        mx = m0; sm = s0;
    }
    __syncthreads();
    if (j < NC) out[b * NC + j] = lg[j] - mx - __logf(sm);
}

// ---------------------------------------------------------------------------
// Host orchestration
// ---------------------------------------------------------------------------
static inline void gemm(const bf16_t* A1, int lda1, int K1,
                        const bf16_t* A2, int lda2, int K2,
                        const bf16_t* Bw, int ldb, int nPad,
                        float* C, int ldc, const float* bias,
                        int M, int N, int act, hipStream_t st)
{
    int Mt = (M + 15) / 16, Nt4 = (N + 63) / 64;
    dim3 blk(32, 4), grd(((Mt + 3) / 4) * Nt4);
    k_gemm_bf16<<<grd, blk, 0, st>>>(A1, lda1, K1, A2, lda2, K2,
                                     Bw, ldb, nPad, C, ldc, bias, M, N, Mt, Nt4, act);
}

#define EWL(n) <<<dim3(((n)+255)/256), dim3(256), 0, stream>>>
#define MEMSET0(p, bytes) (void)hipMemsetAsync((p), 0, (bytes), stream)

extern "C" void kernel_launch(void* const* d_in, const int* in_sizes, int n_in,
                              void* d_out, int out_size, void* d_ws, size_t ws_size,
                              hipStream_t stream)
{
    // ---- inputs in setup_inputs() dict order ----
    const int*   sents      = (const int*)  d_in[0];
    const int*   lengths    = (const int*)  d_in[1];
    const float* emb        = (const float*)d_in[2];
    const float* lstm_Wih_f = (const float*)d_in[3];
    const float* lstm_Whh_f = (const float*)d_in[4];
    const float* lstm_bih_f = (const float*)d_in[5];
    const float* lstm_bhh_f = (const float*)d_in[6];
    const float* gru_Wih_f  = (const float*)d_in[7];
    const float* gru_Whh_f  = (const float*)d_in[8];
    const float* gru_bih_f  = (const float*)d_in[9];
    const float* gru_bhh_f  = (const float*)d_in[10];
    const float* lstm_Wih_b = (const float*)d_in[11];
    const float* lstm_Whh_b = (const float*)d_in[12];
    const float* lstm_bih_b = (const float*)d_in[13];
    const float* lstm_bhh_b = (const float*)d_in[14];
    const float* gru_Wih_b  = (const float*)d_in[15];
    const float* gru_Whh_b  = (const float*)d_in[16];
    const float* gru_bih_b  = (const float*)d_in[17];
    const float* gru_bhh_b  = (const float*)d_in[18];
    const float* lin1_W     = (const float*)d_in[19];
    const float* lin1_b     = (const float*)d_in[20];
    const float* att_Wr     = (const float*)d_in[21];
    const float* att_Ur     = (const float*)d_in[22];
    const float* att_W      = (const float*)d_in[23];
    const float* att_U      = (const float*)d_in[24];
    const float* att_br     = (const float*)d_in[25];
    const float* att_bur    = (const float*)d_in[26];
    const float* att_bw     = (const float*)d_in[27];
    const float* att_bu     = (const float*)d_in[28];
    const float* cls_W      = (const float*)d_in[29];
    const float* cls_b      = (const float*)d_in[30];

    float* outPred = (float*)d_out;            // [BB, NC]
    float* outAttn = outPred + BB * NC;        // [NHOPS, QQ, W1]

    // ---- workspace carve-out ----
    char* wsb = (char*)d_ws; size_t cur = 0;
    auto alloc = [&](size_t bytes) -> void* {
        void* p = wsb + cur; cur = (cur + bytes + 255) & ~(size_t)255; return p;
    };
    bf16_t* Xbf   = (bf16_t*)alloc((size_t)SS * BB * DP * 2);
    bf16_t* OUTbf = (bf16_t*)alloc((size_t)SS * BB * CAT * 2);
    bf16_t* WcF   = (bf16_t*)alloc((size_t)G4 * KC2 * 2);   // [Wih_f | Whh_f]
    bf16_t* WcB   = (bf16_t*)alloc((size_t)G4 * KC2 * 2);   // [Wih_b | Whh_b]
    float*  biasF = (float*) alloc((size_t)G4 * 4);
    float*  biasB = (float*) alloc((size_t)G4 * 4);
    bf16_t* gWihF = (bf16_t*)alloc((size_t)G3P * DAP * 2);
    bf16_t* gWhhF = (bf16_t*)alloc((size_t)G3P * DAP * 2);
    bf16_t* gWihB = (bf16_t*)alloc((size_t)G3P * DAP * 2);
    bf16_t* gWhhB = (bf16_t*)alloc((size_t)G3P * DAP * 2);
    bf16_t* lin1Wb= (bf16_t*)alloc((size_t)LIN_NP * CAT * 2);
    bf16_t* aWr   = (bf16_t*)alloc((size_t)NHOPS * ATT_NP * DAP * 2);
    bf16_t* aUr   = (bf16_t*)alloc((size_t)NHOPS * ATT_NP * DAP * 2);
    bf16_t* aW    = (bf16_t*)alloc((size_t)NHOPS * ATT_NP * DAP * 2);
    bf16_t* aU    = (bf16_t*)alloc((size_t)NHOPS * ATT_NP * DAP * 2);
    bf16_t* mp    = (bf16_t*)alloc((size_t)BB * CAT * 2);
    float*  s_utt = (float*) alloc((size_t)BB * DA * 4);
    bf16_t* btbf  = (bf16_t*)alloc((size_t)W1 * QQ * DAP * 2);
    float*  Gbuf  = (float*) alloc((size_t)BB * G4 * 4);
    float*  h_l   = (float*) alloc((size_t)BB * HH * 4);
    float*  c_l   = (float*) alloc((size_t)BB * HH * 4);
    bf16_t* hbf_l = (bf16_t*)alloc((size_t)BB * DP * 2);
    float*  xg_g  = (float*) alloc((size_t)QQ * G3 * 4);
    float*  hg_g  = (float*) alloc((size_t)QQ * G3 * 4);
    float*  h_g   = (float*) alloc((size_t)QQ * DA * 4);
    bf16_t* hbf_g = (bf16_t*)alloc((size_t)QQ * DAP * 2);
    float*  memF  = (float*) alloc((size_t)W1 * QQ * DA * 4);
    float*  memB  = (float*) alloc((size_t)W1 * QQ * DA * 4);
    float*  memTm = (float*) alloc((size_t)W1 * QQ * DA * 4);
    bf16_t* membf = (bf16_t*)alloc((size_t)W1 * QQ * DAP * 2);
    float*  crb   = (float*) alloc((size_t)W1 * QQ * DA * 4);
    float*  cwb   = (float*) alloc((size_t)W1 * QQ * DA * 4);
    float*  eps   = (float*) alloc((size_t)QQ * DA * 4);
    float*  h_a   = (float*) alloc((size_t)QQ * DA * 4);
    bf16_t* hbf_a = (bf16_t*)alloc((size_t)QQ * DAP * 2);
    float*  hrb   = (float*) alloc((size_t)QQ * DA * 4);
    float*  hub   = (float*) alloc((size_t)QQ * DA * 4);
    (void)ws_size; (void)in_sizes; (void)n_in; (void)out_size;

    // ---- weight casts (K-concatenated LSTM weights) / bias fusion ----
    k_cast_pad EWL(G4*DP)(lstm_Wih_f, WcF, G4, DD, G4, DP, KC2, 0);
    k_cast_pad EWL(G4*DP)(lstm_Whh_f, WcF, G4, HH, G4, DP, KC2, DP);
    k_cast_pad EWL(G4*DP)(lstm_Wih_b, WcB, G4, DD, G4, DP, KC2, 0);
    k_cast_pad EWL(G4*DP)(lstm_Whh_b, WcB, G4, HH, G4, DP, KC2, DP);
    k_vec_add  EWL(G4)(lstm_bih_f, lstm_bhh_f, biasF, G4);
    k_vec_add  EWL(G4)(lstm_bih_b, lstm_bhh_b, biasB, G4);
    k_cast_pad EWL(G3P*DAP)(gru_Wih_f, gWihF, G3, DA, G3P, DAP, DAP, 0);
    k_cast_pad EWL(G3P*DAP)(gru_Whh_f, gWhhF, G3, DA, G3P, DAP, DAP, 0);
    k_cast_pad EWL(G3P*DAP)(gru_Wih_b, gWihB, G3, DA, G3P, DAP, DAP, 0);
    k_cast_pad EWL(G3P*DAP)(gru_Whh_b, gWhhB, G3, DA, G3P, DAP, DAP, 0);
    k_cast_pad EWL(LIN_NP*CAT)(lin1_W, lin1Wb, DA, 600, LIN_NP, CAT, CAT, 0);
    for (int hp = 0; hp < NHOPS; ++hp) {
        k_cast_pad EWL(ATT_NP*DAP)(att_Wr + hp*DA*DA, aWr + hp*ATT_NP*DAP, DA, DA, ATT_NP, DAP, DAP, 0);
        k_cast_pad EWL(ATT_NP*DAP)(att_Ur + hp*DA*DA, aUr + hp*ATT_NP*DAP, DA, DA, ATT_NP, DAP, DAP, 0);
        k_cast_pad EWL(ATT_NP*DAP)(att_W  + hp*DA*DA, aW  + hp*ATT_NP*DAP, DA, DA, ATT_NP, DAP, DAP, 0);
        k_cast_pad EWL(ATT_NP*DAP)(att_U  + hp*DA*DA, aU  + hp*ATT_NP*DAP, DA, DA, ATT_NP, DAP, DAP, 0);
    }

    // ---- embedding gather (time-major, bf16, K-padded) ----
    k_gather_embed EWL(SS*BB*DP)(sents, emb, Xbf);

    // ---- BiLSTM: per step ONE fused dual-source GEMM + gates ----
    for (int dir = 0; dir < 2; ++dir) {
        const bf16_t* Wc   = dir ? WcB : WcF;
        const float*  bcmb = dir ? biasB : biasF;
        MEMSET0(h_l, (size_t)BB*HH*4);
        MEMSET0(c_l, (size_t)BB*HH*4);
        MEMSET0(hbf_l, (size_t)BB*DP*2);
        for (int t = 0; t < SS; ++t) {
            int s = dir ? (SS - 1 - t) : t;
            gemm(Xbf + (size_t)s*BB*DP, DP, DP,        // A1 = x_t, K1 = 320
                 hbf_l, DP, DP,                        // A2 = h,   K2 = 320
                 Wc, KC2, G4, Gbuf, G4, bcmb,
                 BB, G4, /*act=*/0, stream);
            k_lstm_point EWL(BB*DP)(Gbuf, h_l, c_l, hbf_l, OUTbf, lengths,
                                    s, dir ? HH : 0);
        }
    }

    // ---- maxpool over time + lin1 (tanh) ----
    k_maxpool EWL(BB*CAT)(OUTbf, mp);
    gemm(mp, CAT, CAT, mp, CAT, 0, lin1Wb, CAT, LIN_NP,
         s_utt, DA, lin1_b, BB, DA, /*tanh*/1, stream);

    // ---- context windows + BiGRU memory ----
    k_build_bt EWL(W1*QQ*DAP)(s_utt, btbf);
    for (int dir = 0; dir < 2; ++dir) {
        const bf16_t* gWih = dir ? gWihB : gWihF;
        const bf16_t* gWhh = dir ? gWhhB : gWhhF;
        const float*  bih  = dir ? gru_bih_b : gru_bih_f;
        const float*  bhh  = dir ? gru_bhh_b : gru_bhh_f;
        float* memOut = dir ? memB : memF;
        MEMSET0(h_g, (size_t)QQ*DA*4);
        MEMSET0(hbf_g, (size_t)QQ*DAP*2);
        for (int t = 0; t < W1; ++t) {
            int kt = dir ? (W1 - 1 - t) : t;
            gemm(btbf + (size_t)kt*QQ*DAP, DAP, DAP, btbf, DAP, 0,
                 gWih, DAP, G3P, xg_g, G3, bih, QQ, G3, 0, stream);
            gemm(hbf_g, DAP, DAP, hbf_g, DAP, 0,
                 gWhh, DAP, G3P, hg_g, G3, bhh, QQ, G3, 0, stream);
            k_gru_point EWL(QQ*DAP)(xg_g, hg_g, h_g, hbf_g, memOut + (size_t)kt*QQ*DA);
        }
    }
    k_mem_combine EWL(W1*QQ*DAP)(s_utt, memF, memB, memTm, membf);

    // ---- attention hops ----
    k_copy EWL(QQ*DA)(s_utt + DA, eps, QQ*DA);   // eps = s_utt[1:]
    for (int hp = 0; hp < NHOPS; ++hp) {
        gemm(membf, DAP, DAP, membf, DAP, 0, aWr + hp*ATT_NP*DAP, DAP, ATT_NP,
             crb, DA, att_br + hp*DA, W1*QQ, DA, 0, stream);
        gemm(membf, DAP, DAP, membf, DAP, 0, aW + hp*ATT_NP*DAP, DAP, ATT_NP,
             cwb, DA, att_bw + hp*DA, W1*QQ, DA, 0, stream);
        float* scores = outAttn + (size_t)hp*QQ*W1;
        k_attn_softmax<<<dim3(QQ), dim3(64), 0, stream>>>(eps, memTm, scores);
        MEMSET0(h_a, (size_t)QQ*DA*4);
        MEMSET0(hbf_a, (size_t)QQ*DAP*2);
        for (int t = 0; t < W1; ++t) {
            gemm(hbf_a, DAP, DAP, hbf_a, DAP, 0, aUr + hp*ATT_NP*DAP, DAP, ATT_NP,
                 hrb, DA, att_bur + hp*DA, QQ, DA, 0, stream);
            gemm(hbf_a, DAP, DAP, hbf_a, DAP, 0, aU + hp*ATT_NP*DAP, DAP, ATT_NP,
                 hub, DA, att_bu + hp*DA, QQ, DA, 0, stream);
            k_attgru_point EWL(QQ*DAP)(crb, cwb, hrb, hub, scores, h_a, hbf_a, t);
        }
        k_eps_add EWL(QQ*DA)(eps, h_a, QQ*DA);
    }

    // ---- classifier (log_softmax) ----
    k_classify<<<dim3(BB), dim3(32), 0, stream>>>(s_utt, eps, cls_W, cls_b, outPred);
}
